// BasicBlockGenerator_60627758350824
// MI455X (gfx1250) — matrix operand order repaired and checked by reading, not compiled, not run on hardware
//
#include <hip/hip_runtime.h>
#include <hip/hip_bf16.h>

// ---------------------------------------------------------------------------
// Problem constants (match reference)
// ---------------------------------------------------------------------------
#define VP   16384      // V_PREV
#define VN   65536      // V_NEXT (== NNZ_UP, up_row == arange -> pure gather)
#define FIN  512
#define FOUT 256
#define NE   524288
#define BN_EPS 1e-5f

typedef __attribute__((ext_vector_type(16))) __bf16 v16bf;
typedef __attribute__((ext_vector_type(8)))  __bf16 v8bf;
typedef __attribute__((ext_vector_type(4)))  __bf16 v4bf;
typedef __attribute__((ext_vector_type(8)))  float  v8f;

// ---------------------------------------------------------------------------
// Utility: zero a float buffer
// ---------------------------------------------------------------------------
__global__ void zero_kernel(float* p, int n) {
    int i = blockIdx.x * blockDim.x + threadIdx.x;
    if (i < n) p[i] = 0.0f;
}

// fp32 -> bf16 bulk convert (float4 in, 4x bf16 out)
__global__ void f32_to_bf16_kernel(const float* __restrict__ X,
                                   __bf16* __restrict__ Y, size_t n4) {
    size_t stride = (size_t)gridDim.x * blockDim.x;
    for (size_t idx = (size_t)blockIdx.x * blockDim.x + threadIdx.x;
         idx < n4; idx += stride) {
        float4 v = ((const float4*)X)[idx];
        v4bf o = { (__bf16)v.x, (__bf16)v.y, (__bf16)v.z, (__bf16)v.w };
        ((v4bf*)Y)[idx] = o;
    }
}

// ---------------------------------------------------------------------------
// BatchNorm stats: per-channel sum & sum-of-squares, partial sums + atomics.
// ---------------------------------------------------------------------------
__global__ void colsum_kernel(const float* __restrict__ X, int M, int C,
                              int rowsPerBlock,
                              float* __restrict__ sums, float* __restrict__ sumsq) {
    int r0 = blockIdx.x * rowsPerBlock;
    int r1 = r0 + rowsPerBlock; if (r1 > M) r1 = M;
    for (int c = threadIdx.x; c < C; c += blockDim.x) {
        float s = 0.0f, s2 = 0.0f;
        for (int r = r0; r < r1; ++r) {
            float v = X[(size_t)r * C + c];
            s += v; s2 += v * v;
        }
        atomicAdd(&sums[c], s);
        atomicAdd(&sumsq[c], s2);
    }
}

// Fold BN affine into y = a*x + b:  a = g*rstd, b = beta - mu*a
__global__ void bn_finalize_kernel(const float* __restrict__ sums,
                                   const float* __restrict__ sumsq,
                                   int M, int C,
                                   const float* __restrict__ g,
                                   const float* __restrict__ beta,
                                   float* __restrict__ a, float* __restrict__ b) {
    int c = blockIdx.x * blockDim.x + threadIdx.x;
    if (c >= C) return;
    float invM = 1.0f / (float)M;
    float mu   = sums[c] * invM;
    float var  = sumsq[c] * invM - mu * mu;
    float rstd = rsqrtf(var + BN_EPS);
    float ga   = g[c] * rstd;
    a[c] = ga;
    b[c] = beta[c] - mu * ga;
}

// y = relu(a*x + b), fp32 in -> bf16 out (feeds WMMA A operand directly)
__global__ void bn_apply_relu_bf16_kernel(const float* __restrict__ X,
                                          const float* __restrict__ a,
                                          const float* __restrict__ b,
                                          __bf16* __restrict__ Y, int M, int C) {
    const int cv = C >> 2;
    size_t total  = (size_t)M * cv;
    size_t stride = (size_t)gridDim.x * blockDim.x;
    for (size_t idx = (size_t)blockIdx.x * blockDim.x + threadIdx.x;
         idx < total; idx += stride) {
        int c4 = (int)(idx % cv);
        float4 xv = ((const float4*)X)[idx];
        float4 av = ((const float4*)a)[c4];
        float4 bv = ((const float4*)b)[c4];
        float r0 = av.x * xv.x + bv.x; r0 = r0 > 0.0f ? r0 : 0.0f;
        float r1 = av.y * xv.y + bv.y; r1 = r1 > 0.0f ? r1 : 0.0f;
        float r2 = av.z * xv.z + bv.z; r2 = r2 > 0.0f ? r2 : 0.0f;
        float r3 = av.w * xv.w + bv.w; r3 = r3 > 0.0f ? r3 : 0.0f;
        v4bf o = { (__bf16)r0, (__bf16)r1, (__bf16)r2, (__bf16)r3 };
        ((v4bf*)Y)[idx] = o;
    }
}

// ---------------------------------------------------------------------------
// Pack W[K,N] (fp32, row-major) into bf16 B-fragments in contiguous k order (matches load_afrag):
//   fragment f = kt*(N/16) + t16n ; per lane 16 contiguous bf16:
//   elem i -> K = kt*32 + (lane>>4)*16 + i, col = t16n*16 + (lane&15)
// One thread per (fragment, lane). Done once per GEMM; weights are tiny.
// ---------------------------------------------------------------------------
__global__ void pack_w_kernel(const float* __restrict__ W,
                              __bf16* __restrict__ PB, int K, int N) {
    int tid = blockIdx.x * blockDim.x + threadIdx.x;
    int total = (K >> 5) * (N >> 4) * 32;
    if (tid >= total) return;
    int lane = tid & 31;
    int frag = tid >> 5;
    int t16n = frag % (N >> 4);
    int kt   = frag / (N >> 4);
    int col   = t16n * 16 + (lane & 15);
    int kbase = kt * 32 + (lane >> 4) * 16;
    __bf16* outp = PB + (size_t)tid * 16;
    #pragma unroll
    for (int i = 0; i < 16; ++i)
        outp[i] = (__bf16)W[(size_t)(kbase + i) * N + col];
}

// A fragment loader: bf16 row-major A; two contiguous 16-byte runs.
//   elem i(0..15) -> K = k0 + half*16 + i
__device__ inline v16bf load_afrag(const __bf16* p) {
    v8bf lo = *(const v8bf*)p;
    v8bf hi = *(const v8bf*)(p + 8);
    v16bf r;
    #pragma unroll
    for (int i = 0; i < 8; ++i) { r[i] = lo[i]; r[8 + i] = hi[i]; }
    return r;
}

__device__ inline v8f wmma_bf16(v16bf a, v16bf b, v8f c) {
    return __builtin_amdgcn_wmma_f32_16x16x32_bf16(
        /*neg_a=*/false, a, /*neg_b=*/false, b,
        /*c_mod=*/(short)0, c, /*reuse_a=*/false, /*reuse_b=*/false);
}

// ---------------------------------------------------------------------------
// WMMA GEMM:  C[M,N] (+)= A_bf16[M,K] * Wpacked[K,N] (+ bias[N])
// One wave computes a 32x32 output block = 2x2 WMMA tiles (4 v_wmma / K-step,
// 2x operand reuse). ACCUM/HASBIAS are compile-time -> branch-free epilogue.
// Wave-uniform early-exit keeps EXEC all-1s (WMMA requirement).
// ---------------------------------------------------------------------------
template<bool ACCUM, bool HASBIAS>
__global__ void wmma_gemm_bf16_kernel(const __bf16* __restrict__ A,
                                      const __bf16* __restrict__ PB,
                                      const float* __restrict__ bias,
                                      float* __restrict__ C,
                                      int M, int N, int K) {
    const int lane = threadIdx.x & 31;
    const int wave = threadIdx.x >> 5;
    const int tilesN = N >> 5;                       // 32-wide blocks
    const int numTiles = (M >> 5) * tilesN;
    const int tile = blockIdx.x * (blockDim.x >> 5) + wave;
    if (tile >= numTiles) return;

    const int tm = tile / tilesN;
    const int tn = tile - tm * tilesN;
    const int half = lane >> 4;
    const int l15  = lane & 15;
    const int fragsPerK = N >> 4;                    // 16-col fragments per K-step

    const __bf16* a0p = A + (size_t)(tm * 32 + l15) * K + half * 16;
    const __bf16* a1p = a0p + (size_t)16 * K;
    const __bf16* bbase = PB + ((size_t)(tn * 2) * 32 + lane) * 16;
    const size_t  bstep = (size_t)fragsPerK * 32 * 16;   // next K-step

    v8f acc00 = {}, acc01 = {}, acc10 = {}, acc11 = {};
    for (int kt = 0; kt < (K >> 5); ++kt) {
        v16bf a0 = load_afrag(a0p + kt * 32);
        v16bf a1 = load_afrag(a1p + kt * 32);
        v16bf b0 = *(const v16bf*)(bbase + (size_t)kt * bstep);
        v16bf b1 = *(const v16bf*)(bbase + (size_t)kt * bstep + 32 * 16);
        acc00 = wmma_bf16(a0, b0, acc00);
        acc01 = wmma_bf16(a0, b1, acc01);
        acc10 = wmma_bf16(a1, b0, acc10);
        acc11 = wmma_bf16(a1, b1, acc11);
    }

    const int c0 = tn * 32 + l15;
    const int c1 = c0 + 16;
    const float bv0 = HASBIAS ? bias[c0] : 0.0f;
    const float bv1 = HASBIAS ? bias[c1] : 0.0f;
    float* r0 = C + (size_t)(tm * 32 + half * 8) * N;
    float* r1 = r0 + (size_t)16 * N;
    #pragma unroll
    for (int r = 0; r < 8; ++r) {
        size_t o = (size_t)r * N;
        float v00 = acc00[r] + bv0, v01 = acc01[r] + bv1;
        float v10 = acc10[r] + bv0, v11 = acc11[r] + bv1;
        if (ACCUM) {
            r0[o + c0] += v00; r0[o + c1] += v01;
            r1[o + c0] += v10; r1[o + c1] += v11;
        } else {
            r0[o + c0] = v00; r0[o + c1] = v01;
            r1[o + c0] = v10; r1[o + c1] = v11;
        }
    }
}

// ---------------------------------------------------------------------------
// Gather-with-scale (the degenerate spmm: up_row == arange):
//   D[i, :] = val[i] * S[col[i], :] + bias[:]
// ---------------------------------------------------------------------------
__global__ void gather_scale_bias_kernel(const float* __restrict__ S,
                                         const int*   __restrict__ col,
                                         const float* __restrict__ val,
                                         const float* __restrict__ bias,
                                         float* __restrict__ D,
                                         int Mn, int Cn) {
    const int cv = Cn >> 2;
    size_t total  = (size_t)Mn * cv;
    size_t stride = (size_t)gridDim.x * blockDim.x;
    for (size_t idx = (size_t)blockIdx.x * blockDim.x + threadIdx.x;
         idx < total; idx += stride) {
        int i  = (int)(idx / cv);
        int c4 = (int)(idx % cv);
        float v = val[i];
        float4 s = ((const float4*)(S + (size_t)col[i] * Cn))[c4];
        float4 b = ((const float4*)bias)[c4];
        float4 d;
        d.x = v * s.x + b.x; d.y = v * s.y + b.y;
        d.z = v * s.z + b.z; d.w = v * s.w + b.w;
        ((float4*)(D + (size_t)i * Cn))[c4] = d;
    }
}

// ---------------------------------------------------------------------------
// Edge scatter:  OUT[dst[e], :] += w[e] * UJ[src[e], :]
// 64 consecutive threads per edge (float4 loads + fp32 global atomics).
// ---------------------------------------------------------------------------
__global__ void edge_scatter_kernel(const float* __restrict__ UJ,
                                    const int*   __restrict__ src,
                                    const int*   __restrict__ dst,
                                    const float* __restrict__ w,
                                    float* __restrict__ OUT,
                                    int E, int Cn) {
    const int cv = Cn >> 2;
    size_t total  = (size_t)E * cv;
    size_t stride = (size_t)gridDim.x * blockDim.x;
    for (size_t idx = (size_t)blockIdx.x * blockDim.x + threadIdx.x;
         idx < total; idx += stride) {
        int e  = (int)(idx / cv);
        int c4 = (int)(idx % cv);
        float we = w[e];
        float4 u = ((const float4*)(UJ + (size_t)src[e] * Cn))[c4];
        float* o = OUT + (size_t)dst[e] * Cn + (size_t)c4 * 4;
        atomicAdd(o + 0, we * u.x);
        atomicAdd(o + 1, we * u.y);
        atomicAdd(o + 2, we * u.z);
        atomicAdd(o + 3, we * u.w);
    }
}

// ---------------------------------------------------------------------------
// Host launcher
// ---------------------------------------------------------------------------
extern "C" void kernel_launch(void* const* d_in, const int* in_sizes, int n_in,
                              void* d_out, int out_size, void* d_ws, size_t ws_size,
                              hipStream_t stream) {
    const float* x        = (const float*)d_in[0];
    /* up_row (d_in[1]) == arange(VN): spmm degenerates to a gather; unused */
    const int*   up_col   = (const int*)  d_in[2];
    const float* up_val   = (const float*)d_in[3];
    const int*   e_src    = (const int*)  d_in[4];
    const int*   e_dst    = (const int*)  d_in[5];
    const float* adj_val  = (const float*)d_in[6];
    const float* W_res    = (const float*)d_in[7];
    const float* b_res    = (const float*)d_in[8];
    const float* bn0_g    = (const float*)d_in[9];
    const float* bn0_b    = (const float*)d_in[10];
    const float* bn1_g    = (const float*)d_in[11];
    const float* bn1_b    = (const float*)d_in[12];
    const float* Ui0_W    = (const float*)d_in[13];
    const float* Ui0_b    = (const float*)d_in[14];
    const float* Uj0_W    = (const float*)d_in[15];
    const float* Uj0_b    = (const float*)d_in[16];
    const float* Ui1_W    = (const float*)d_in[17];
    const float* Ui1_b    = (const float*)d_in[18];
    const float* Uj1_W    = (const float*)d_in[19];
    const float* Uj1_b    = (const float*)d_in[20];
    float* out = (float*)d_out;

    // ---- workspace layout (bytes, 256B-aligned regions) ----
    char* base = (char*)d_ws;
    float* sum0 = (float*)(base + 0);       // 512
    float* sq0  = (float*)(base + 2048);    // 512
    float* a0   = (float*)(base + 4096);    // 512
    float* bb0  = (float*)(base + 6144);    // 512
    float* sum1 = (float*)(base + 8192);    // 256
    float* sq1  = (float*)(base + 9216);    // 256
    float* a1   = (float*)(base + 10240);   // 256
    float* bb1  = (float*)(base + 11264);   // 256
    size_t o = 32768;
    __bf16* xb   = (__bf16*)(base + o); o += (size_t)VP * FIN * 2;   // x as bf16
    __bf16* xn   = (__bf16*)(base + o); o += (size_t)VP * FIN * 2;   // relu(bn0(x)) bf16
    __bf16* Wr_p = (__bf16*)(base + o); o += (size_t)FIN * FOUT * 2; // packed W_res
    __bf16* Ui0p = (__bf16*)(base + o); o += (size_t)FIN * FOUT * 2;
    __bf16* Uj0p = (__bf16*)(base + o); o += (size_t)FIN * FOUT * 2;
    __bf16* Ui1p = (__bf16*)(base + o); o += (size_t)FOUT * FOUT * 2;
    __bf16* Uj1p = (__bf16*)(base + o); o += (size_t)FOUT * FOUT * 2;
    float* Hres  = (float*)(base + o); o += (size_t)VP * FOUT * 4;   // x @ W_res
    float* P     = (float*)(base + o); o += (size_t)VP * FOUT * 4;   // xn @ Ui0
    float* Q     = (float*)(base + o); o += (size_t)VP * FOUT * 4;   // xn @ Uj0
    float* t1    = (float*)(base + o); o += (size_t)VN * FOUT * 4;   // ECC0 output
    float* ujg   = (float*)(base + o); o += (size_t)VN * FOUT * 4;   // Uj0 messages
    __bf16* t1n  = xb;   // overlay: xb+xn (32MB) dead by then; need 32MB exactly
    float*  uj1  = ujg;  // overlay: ujg dead after first scatter

    const int T = 256;

    // 1) zero BN accumulators
    zero_kernel<<<(3072 + T - 1) / T, T, 0, stream>>>(sum0, 3072);

    // 2) precision staging: x -> bf16 ; pack all 5 weight matrices
    f32_to_bf16_kernel<<<8192, T, 0, stream>>>(x, xb, (size_t)VP * FIN / 4);
    pack_w_kernel<<<((FIN/32)*(FOUT/16)*32 + T - 1) / T, T, 0, stream>>>(W_res, Wr_p, FIN, FOUT);
    pack_w_kernel<<<((FIN/32)*(FOUT/16)*32 + T - 1) / T, T, 0, stream>>>(Ui0_W, Ui0p, FIN, FOUT);
    pack_w_kernel<<<((FIN/32)*(FOUT/16)*32 + T - 1) / T, T, 0, stream>>>(Uj0_W, Uj0p, FIN, FOUT);
    pack_w_kernel<<<((FOUT/32)*(FOUT/16)*32 + T - 1) / T, T, 0, stream>>>(Ui1_W, Ui1p, FOUT, FOUT);
    pack_w_kernel<<<((FOUT/32)*(FOUT/16)*32 + T - 1) / T, T, 0, stream>>>(Uj1_W, Uj1p, FOUT, FOUT);

    // 3) BN0 over x [VP, FIN], then xn = relu(bn0(x)) as bf16
    colsum_kernel<<<VP / 128, T, 0, stream>>>(x, VP, FIN, 128, sum0, sq0);
    bn_finalize_kernel<<<(FIN + T - 1) / T, T, 0, stream>>>(sum0, sq0, VP, FIN,
                                                            bn0_g, bn0_b, a0, bb0);
    bn_apply_relu_bf16_kernel<<<4096, T, 0, stream>>>(x, a0, bb0, xn, VP, FIN);

    // 4) dense contractions BEFORE the gather (gather commutes with GEMM):
    //    3x [16384,512]x[512,256] instead of [65536,...] -> 4x fewer FLOPs
    {
        int tiles = (VP / 32) * (FOUT / 32);
        wmma_gemm_bf16_kernel<false, false><<<(tiles + 7) / 8, T, 0, stream>>>(xb, Wr_p, nullptr, Hres, VP, FOUT, FIN);
        wmma_gemm_bf16_kernel<false, false><<<(tiles + 7) / 8, T, 0, stream>>>(xn, Ui0p, nullptr, P,    VP, FOUT, FIN);
        wmma_gemm_bf16_kernel<false, false><<<(tiles + 7) / 8, T, 0, stream>>>(xn, Uj0p, nullptr, Q,    VP, FOUT, FIN);
    }

    // 5) gathers (up_row == arange -> gather with scale), biases folded in
    gather_scale_bias_kernel<<<8192, T, 0, stream>>>(Hres, up_col, up_val, b_res, out, VN, FOUT);
    gather_scale_bias_kernel<<<8192, T, 0, stream>>>(P,    up_col, up_val, Ui0_b, t1,  VN, FOUT);
    gather_scale_bias_kernel<<<8192, T, 0, stream>>>(Q,    up_col, up_val, Uj0_b, ujg, VN, FOUT);

    // 6) ECC0 edge aggregation: t1[dst] += adj_val * ujg[src]
    edge_scatter_kernel<<<8192, T, 0, stream>>>(ujg, e_src, e_dst, adj_val, t1, NE, FOUT);

    // 7) BN1 over t1 [VN, FOUT] -> t1n (bf16, feeds ECC1 GEMMs)
    colsum_kernel<<<VN / 128, T, 0, stream>>>(t1, VN, FOUT, 128, sum1, sq1);
    bn_finalize_kernel<<<(FOUT + T - 1) / T, T, 0, stream>>>(sum1, sq1, VN, FOUT,
                                                             bn1_g, bn1_b, a1, bb1);
    bn_apply_relu_bf16_kernel<<<8192, T, 0, stream>>>(t1, a1, bb1, t1n, VN, FOUT);

    // 8) ECC1 dense parts [65536,256]x[256,256]
    {
        int tiles = (VN / 32) * (FOUT / 32);
        // d_out += t1n @ Ui1_W + Ui1_b  (accumulates onto residual already in d_out)
        wmma_gemm_bf16_kernel<true,  true><<<(tiles + 7) / 8, T, 0, stream>>>(t1n, Ui1p, Ui1_b, out, VN, FOUT, FOUT);
        // uj1 = t1n @ Uj1_W + Uj1_b
        wmma_gemm_bf16_kernel<false, true><<<(tiles + 7) / 8, T, 0, stream>>>(t1n, Uj1p, Uj1_b, uj1, VN, FOUT, FOUT);
    }

    // 9) ECC1 edge aggregation directly into the output
    edge_scatter_kernel<<<8192, T, 0, stream>>>(uj1, e_src, e_dst, adj_val, out, NE, FOUT);

    (void)in_sizes; (void)n_in; (void)out_size; (void)ws_size;
}
